// WindowAttention_12816182411473
// MI455X (gfx1250) — matrix operand order repaired and checked by reading, not compiled, not run on hardware
//
#include <hip/hip_runtime.h>
#include <hip/hip_bf16.h>

typedef _Float16 half_t;
typedef __attribute__((ext_vector_type(16))) _Float16 v16h;
typedef __attribute__((ext_vector_type(8)))  float    v8f;
typedef unsigned int u32;
typedef __attribute__((ext_vector_type(4))) u32 u32x4;
typedef __attribute__((ext_vector_type(8))) u32 u32x8;

#define WMMA_F32_F16(a, b, c) \
    __builtin_amdgcn_wmma_f32_16x16x32_f16(false, (a), false, (b), (short)0, (c), false, false)

// xor-butterfly lane exchange via ds_swizzle_b32 (group-of-32 mode, no LDS
// memory, no bounds clamping): offset = and_mask | or_mask<<5 | xor_mask<<10.
// xor masks 1/2/4/8 stay inside each 16-lane half (one score row).
#define SWIZZLE_XOR(x, k) \
    __int_as_float(__builtin_amdgcn_ds_swizzle(__float_as_int(x), 0x1f | ((k) << 10)))

// ---------------------------------------------------------------------------
// Tensor Data Mover: 2D tile (tile_d0 contiguous elems x tile_d1 rows) from
// global -> packed LDS.  D# per CDNA5 ISA.
// Descriptor operands are wave-uniform -> SGPR tuples.
// ---------------------------------------------------------------------------
__device__ __forceinline__ void tdm_load_2d(u32 lds_off, const void* gptr,
                                            u32 dsz_code,          // 0=1B,1=2B,2=4B
                                            u32 tile_d0, u32 tile_d1,
                                            u32 tensor_d0, u32 tensor_d1,
                                            u32 stride0 /* elems */) {
    unsigned long long ga = (unsigned long long)gptr;
    u32x4 g0;
    g0[0] = 1u;                                  // count=1, user mode
    g0[1] = lds_off;                             // lds_addr [63:32]
    g0[2] = (u32)ga;                             // global_addr [95:64]
    g0[3] = (u32)(ga >> 32) | (2u << 30);        // global_addr hi | type=2
    u32x8 g1;
    g1[0] = dsz_code << 16;                      // wg_mask=0 | data_size
    g1[1] = (tensor_d0 & 0xffffu) << 16;         // tensor_dim0 lo16
    g1[2] = (tensor_d0 >> 16) | ((tensor_d1 & 0xffffu) << 16);
    g1[3] = (tensor_d1 >> 16) | (tile_d0 << 16); // tensor_dim1 hi | tile_dim0
    g1[4] = tile_d1 & 0xffffu;                   // tile_dim1 | tile_dim2=0
    g1[5] = stride0;                             // tensor_dim0_stride lo32
    g1[6] = 0u;                                  // stride0 hi16 | dim1_stride lo16
    g1[7] = 0u;
    asm volatile("tensor_load_to_lds %0, %1" :: "s"(g0), "s"(g1) : "memory");
}

// ---------------------------------------------------------------------------
// One-shot f32 -> f16 weight conversion (enables verbatim TDM tile copies).
// ---------------------------------------------------------------------------
__global__ __launch_bounds__(256)
void cvt_f32_f16(const float* __restrict__ in, half_t* __restrict__ out, int n) {
    int i = blockIdx.x * 256 + threadIdx.x;
    if (i < n) out[i] = (half_t)in[i];
}

// ---------------------------------------------------------------------------
// One-shot relative-position-bias expansion: biasG[h][n][m], 12x49x49 f32.
// rpi(n,m) = ((n%7 - m%7)+6)*13 + ((n/7 - m/7)+6)   (meshgrid 'xy' order)
// ---------------------------------------------------------------------------
__global__ __launch_bounds__(256)
void build_bias(const float* __restrict__ rpb, float* __restrict__ biasG) {
    int i = blockIdx.x * 256 + threadIdx.x;          // over 12*2401
    if (i < 12 * 2401) {
        int h  = i / 2401;
        int nm = i - h * 2401;
        int n  = nm / 49;
        int m  = nm - n * 49;
        int rp = ((n % 7 - m % 7) + 6) * 13 + ((n / 7 - m / 7) + 6);
        biasG[i] = rpb[rp * 12 + h];
    }
}

// ---------------------------------------------------------------------------
// Tiled GEMM: C[M,Nw] = A[M,K] * B16[K,Nw]; K mult of 32, M,Nw mult of 64.
// 128 threads (4 waves); wave = 16-row stripe of the 64x64 block tile.
// Double-buffered LDS: TDM for tile k+1 is issued before computing tile k,
// so the DMA runs under the WMMAs; s_wait_tensorcnt lands after compute.
// ---------------------------------------------------------------------------
template <typename AT, typename OT>
__global__ __launch_bounds__(128)
void gemm64x64_wmma_f16(const AT* __restrict__ A, const half_t* __restrict__ B16,
                        OT* __restrict__ C, int K, int Nw) {
    __shared__ half_t As[2][64][32];    // [m][k], packed (TDM write order)
    __shared__ half_t Bs[2][32][64];    // [k][n], packed (TDM write order)

    const int tid  = threadIdx.x;
    const int lane = tid & 31;
    const int wave = tid >> 5;
    const int rA   = lane & 15;
    const int g    = lane >> 4;
    const int m0   = blockIdx.y * 64;
    const int n0   = blockIdx.x * 64;

    auto stage = [&](int k0, int p) {
        if (wave == 0) {
            // B tile: 32 rows x 64 contiguous halfs, row stride Nw
            tdm_load_2d((u32)(unsigned long long)&Bs[p][0][0],
                        B16 + (size_t)k0 * Nw + n0, /*dsz=*/1,
                        /*tile*/64, 32, /*tensor*/(u32)Nw, (u32)K, (u32)Nw);
            if (sizeof(AT) == 2) {
                // A tile: 64 rows x 32 contiguous halfs, row stride K
                tdm_load_2d((u32)(unsigned long long)&As[p][0][0],
                            (const void*)(A + (size_t)m0 * K + k0), /*dsz=*/1,
                            /*tile*/32, 64, /*tensor*/(u32)K, 0x100000u, (u32)K);
            }
        }
        if (sizeof(AT) == 4) {
            // Manual f32 -> f16 staging of the A tile (all 128 threads)
            #pragma unroll
            for (int i = 0; i < 2; ++i) {
                int chunk = tid + i * 128;           // 256 chunks of 8
                int r  = chunk >> 2;
                int kc = (chunk & 3) * 8;
                const AT* src = A + (size_t)(m0 + r) * K + k0 + kc;
                __builtin_prefetch(src + 32, 0, 0);  // next k-tile
                #pragma unroll
                for (int j = 0; j < 8; ++j) As[p][r][kc + j] = (half_t)src[j];
            }
        }
    };

    v8f acc[4] = {};
    int p = 0;

    stage(0, 0);
    if (wave == 0) __builtin_amdgcn_s_wait_tensorcnt(0);
    __syncthreads();

    for (int k0 = 0; k0 < K; k0 += 32) {
        if (k0 + 32 < K) stage(k0 + 32, p ^ 1);   // prefetch next tile (other buffer)

        // A fragment: row = wave*16+rA, K = (e<8?e:e+8)+8g
        v16h a;
        #pragma unroll
        for (int e = 0; e < 16; ++e) {
            int kk = (e < 8 ? e : e + 8) + 8 * g;
            a[e] = As[p][wave * 16 + rA][kk];
        }
        #pragma unroll
        for (int nt = 0; nt < 4; ++nt) {
            v16h b;
            #pragma unroll
            for (int e = 0; e < 16; ++e)
                b[e] = Bs[p][(e < 8 ? e : e + 8) + 8 * g][nt * 16 + rA];
            acc[nt] = WMMA_F32_F16(a, b, acc[nt]);
        }

        if (wave == 0) __builtin_amdgcn_s_wait_tensorcnt(0);
        __syncthreads();
        p ^= 1;
    }

    // C layout: col = lane&15 (+16*nt), row = r + 8*(lane>>4) (+16*wave)
    #pragma unroll
    for (int nt = 0; nt < 4; ++nt) {
        int col = n0 + nt * 16 + rA;
        #pragma unroll
        for (int r = 0; r < 8; ++r) {
            int row = m0 + wave * 16 + r + 8 * g;
            C[(size_t)row * Nw + col] = (OT)acc[nt][r];
        }
    }
}

// ---------------------------------------------------------------------------
// Fused window attention: one wave per (window, head). 128 threads = 4 heads.
// grid.x = 8192 * 3 (head groups of 4). N=49 padded to 64, hd=32 (one K-step).
// Shift mask + per-head bias slices staged by TDM.
// ---------------------------------------------------------------------------
__global__ __launch_bounds__(128)
void window_attn_wmma(const half_t* __restrict__ qkv, const float* __restrict__ mask,
                      const float* __restrict__ biasG, half_t* __restrict__ out) {
    __shared__ float  maskS[49 * 49];     // shared by the 4 heads of this block
    __shared__ float  biasS[4][49 * 49];  // per-wave head bias slice
    __shared__ half_t probs[4][64][72];   // per-wave softmax probabilities
    __shared__ half_t vT[4][32][72];      // per-wave V transposed: [d][token]

    const int bi   = blockIdx.x;
    const int b    = bi / 3;
    const int hg   = bi - b * 3;
    const int tid  = threadIdx.x;
    const int lane = tid & 31;
    const int wave = tid >> 5;
    const int h    = hg * 4 + wave;
    const int w    = b & 1023;            // mask index = window % nW
    const int rA   = lane & 15;
    const int g    = lane >> 4;

    const size_t rowbase = (size_t)b * 49 * 1152;

    // Stage shift mask (wave 0) and this wave's bias slice via TDM
    if (wave == 0) {
        tdm_load_2d((u32)(unsigned long long)&maskS[0],
                    mask + (size_t)w * 2401, /*dsz=*/2,
                    /*tile*/2401, 1, /*tensor*/2401, 1, 2401);
    }
    tdm_load_2d((u32)(unsigned long long)&biasS[wave][0],
                biasG + (size_t)h * 2401, /*dsz=*/2,
                /*tile*/2401, 1, /*tensor*/2401, 1, 2401);

    // Stage V transposed into this wave's LDS region (zero-pad tokens 49..63)
    for (int m = lane; m < 64; m += 32) {
        if (m < 49) {
            const half_t* src = qkv + rowbase + (size_t)m * 1152 + 768 + h * 32;
            #pragma unroll
            for (int d = 0; d < 32; ++d) vT[wave][d][m] = src[d];
        } else {
            #pragma unroll
            for (int d = 0; d < 32; ++d) vT[wave][d][m] = (half_t)0.f;
        }
    }

    // K fragments (B-matrix layout): b[e] = k[m][e+16g], m = nt*16 + rA
    v16h kf[4];
    #pragma unroll
    for (int nt = 0; nt < 4; ++nt) {
        int m = nt * 16 + rA;
        v16h kv = {};
        if (m < 49) {
            const half_t* src = qkv + rowbase + (size_t)m * 1152 + 384 + h * 32 + 16 * g;
            #pragma unroll
            for (int e = 0; e < 16; ++e) kv[e] = src[e];
        }
        kf[nt] = kv;
    }

    __builtin_amdgcn_s_wait_tensorcnt(0);   // own bias slice (and wave0: mask)
    __syncthreads();                        // mask visible to all waves

    const float scale = 0.17677669529663687f;   // 32^-0.5

    // --- per row-tile: S = q k^T, add scale/bias/mask, softmax, emit probs ---
    #pragma unroll
    for (int mt = 0; mt < 4; ++mt) {
        // Q fragment (A-matrix layout): row = mt*16 + rA, K = e+16g
        v16h qf = {};
        {
            int n = mt * 16 + rA;
            if (n < 49) {
                const half_t* src = qkv + rowbase + (size_t)n * 1152 + h * 32;
                #pragma unroll
                for (int e = 0; e < 16; ++e) {
                    int d = e + 16 * g;
                    qf[e] = src[d];
                }
            }
        }
        v8f s[4];
        #pragma unroll
        for (int nt = 0; nt < 4; ++nt) {
            v8f z = {};
            s[nt] = WMMA_F32_F16(qf, kf[nt], z);
        }
        // softmax: VGPR slot r = one row (n = mt*16 + r + 8g), 16 lanes = cols
        #pragma unroll
        for (int r = 0; r < 8; ++r) {
            int n = mt * 16 + r + 8 * g;
            float vals[4];
            #pragma unroll
            for (int nt = 0; nt < 4; ++nt) {
                int m = nt * 16 + rA;
                float sv = s[nt][r] * scale;
                if (n < 49 && m < 49) {
                    int nm = n * 49 + m;
                    sv += biasS[wave][nm] + maskS[nm];
                } else {
                    sv = -1e30f;
                }
                vals[nt] = sv;
            }
            float mx = fmaxf(fmaxf(vals[0], vals[1]), fmaxf(vals[2], vals[3]));
            mx = fmaxf(mx, SWIZZLE_XOR(mx, 1));
            mx = fmaxf(mx, SWIZZLE_XOR(mx, 2));
            mx = fmaxf(mx, SWIZZLE_XOR(mx, 4));
            mx = fmaxf(mx, SWIZZLE_XOR(mx, 8));
            float sum = 0.f;
            #pragma unroll
            for (int nt = 0; nt < 4; ++nt) { vals[nt] = __expf(vals[nt] - mx); sum += vals[nt]; }
            sum += SWIZZLE_XOR(sum, 1);
            sum += SWIZZLE_XOR(sum, 2);
            sum += SWIZZLE_XOR(sum, 4);
            sum += SWIZZLE_XOR(sum, 8);
            float inv = 1.f / sum;
            #pragma unroll
            for (int nt = 0; nt < 4; ++nt)
                probs[wave][n][nt * 16 + rA] = (half_t)(vals[nt] * inv);
        }
    }

    // --- out = probs @ V : M=64, K=64 (2 steps), N=32 (2 tiles) ---
    #pragma unroll
    for (int mt = 0; mt < 4; ++mt) {
        v16h pa[2];
        #pragma unroll
        for (int ks = 0; ks < 2; ++ks) {
            #pragma unroll
            for (int e = 0; e < 16; ++e) {
                int kk = (e < 8 ? e : e + 8) + 8 * g + ks * 32;
                pa[ks][e] = probs[wave][mt * 16 + rA][kk];
            }
        }
        #pragma unroll
        for (int ntv = 0; ntv < 2; ++ntv) {
            v8f accv = {};
            #pragma unroll
            for (int ks = 0; ks < 2; ++ks) {
                v16h bf;
                #pragma unroll
                for (int e = 0; e < 16; ++e)
                    bf[e] = vT[wave][ntv * 16 + rA][ks * 32 + (e < 8 ? e : e + 8) + 8 * g];
                accv = WMMA_F32_F16(pa[ks], bf, accv);
            }
            int d = ntv * 16 + rA;
            #pragma unroll
            for (int r = 0; r < 8; ++r) {
                int n = mt * 16 + r + 8 * g;
                if (n < 49)
                    out[(size_t)(b * 49 + n) * 384 + h * 32 + d] = (half_t)accv[r];
            }
        }
    }
}

// ---------------------------------------------------------------------------
extern "C" void kernel_launch(void* const* d_in, const int* in_sizes, int n_in,
                              void* d_out, int out_size, void* d_ws, size_t ws_size,
                              hipStream_t stream) {
    (void)in_sizes; (void)n_in; (void)out_size; (void)ws_size;

    const float* x      = (const float*)d_in[0];   // [8192*49, 384]
    const float* mask   = (const float*)d_in[1];   // [1024, 49, 49]
    const float* w_qkv  = (const float*)d_in[2];   // [384, 1152]
    const float* w_proj = (const float*)d_in[3];   // [384, 384]
    const float* rpb    = (const float*)d_in[4];   // [169, 12]
    float*       outp   = (float*)d_out;           // [8192*49, 384]

    const int M = 8192 * 49;                       // 401408, multiple of 64

    half_t* qkv      = (half_t*)d_ws;                        // M x 1152 f16
    half_t* attn_out = qkv + (size_t)M * 1152;               // M x 384  f16
    half_t* wq16     = attn_out + (size_t)M * 384;           // 384x1152 f16
    half_t* wp16     = wq16 + 384 * 1152;                    // 384x384  f16
    float*  biasG    = (float*)(wp16 + 384 * 384);           // 12x49x49 f32

    // 0) one-shot prep: weights f32->f16, bias table expansion
    cvt_f32_f16<<<(384 * 1152 + 255) / 256, 256, 0, stream>>>(w_qkv, wq16, 384 * 1152);
    cvt_f32_f16<<<(384 * 384 + 255) / 256, 256, 0, stream>>>(w_proj, wp16, 384 * 384);
    build_bias<<<(12 * 2401 + 255) / 256, 256, 0, stream>>>(rpb, biasG);

    // 1) qkv = x @ w_qkv  (f32 in, f16 out)
    dim3 g1(1152 / 64, M / 64);
    gemm64x64_wmma_f16<float, half_t><<<g1, 128, 0, stream>>>(x, wq16, qkv, 384, 1152);

    // 2) fused attention (QK^T + bias + mask + softmax + PV)
    window_attn_wmma<<<8192 * 3, 128, 0, stream>>>(qkv, mask, biasG, attn_out);

    // 3) out = attn_out @ w_proj  (f16 in, f32 out)
    dim3 g3(384 / 64, M / 64);
    gemm64x64_wmma_f16<half_t, float><<<g3, 128, 0, stream>>>(attn_out, wp16, outp, 384, 384);
}
